// DeltaNetAttentionCore_70334384439584
// MI455X (gfx1250) — matrix operand order repaired and checked by reading, not compiled, not run on hardware
//
#include <hip/hip_runtime.h>
#include <hip/hip_bf16.h>

// ---------------- constants -------------------------------------------------
#define S_LEN 2048
#define B_SZ  2
#define H_DIM 2048          // hidden
#define C_DIM 2048          // NH*HD
#define NH    16
#define HD    128
#define KCONV 4

typedef __bf16 bf16;
typedef bf16  v16bf __attribute__((ext_vector_type(16)));
typedef float v8f   __attribute__((ext_vector_type(8)));

union FragAB { v16bf v; uint4 q[2]; };

// ---------------- fp32 -> bf16 convert --------------------------------------
__global__ __launch_bounds__(256)
void convert_bf16_kernel(const float* __restrict__ x, bf16* __restrict__ y, int n)
{
    int i = blockIdx.x * 256 + threadIdx.x;
    if (i < n) y[i] = (bf16)x[i];
}

// W [K=2048][N=2048] fp32 -> WT [N][K] bf16 (reads coalesced along N)
__global__ __launch_bounds__(256)
void transpose_convert_kernel(const float* __restrict__ W, bf16* __restrict__ WT)
{
    int idx = blockIdx.x * 256 + threadIdx.x;     // 2048*2048 elements
    int n = idx & 2047;
    int k = idx >> 11;
    WT[(size_t)n * 2048 + k] = (bf16)W[idx];
}

// ---------------- beta = sigmoid(hs @ Wb) -----------------------------------
// hs rows r = s*B + b (natural [s,b,H] order); output beta[b][s][n]
__global__ __launch_bounds__(256)
void beta_kernel(const float* __restrict__ hs, const float* __restrict__ Wb,
                 float* __restrict__ beta)
{
    int r = blockIdx.x;                 // 0..S*B-1
    int n = threadIdx.x >> 4;           // 0..15
    int l = threadIdx.x & 15;
    const float* row = hs + (size_t)r * H_DIM;
    float sum = 0.f;
    for (int h = l; h < H_DIM; h += 16)
        sum += row[h] * Wb[h * NH + n];
#pragma unroll
    for (int off = 8; off > 0; off >>= 1)
        sum += __shfl_xor(sum, off, 16);
    if (l == 0) {
        int s = r >> 1, b = r & 1;
        beta[(size_t)(b * S_LEN + s) * NH + n] = 1.f / (1.f + __expf(-sum));
    }
}

// ---------------- WMMA GEMM (register-blocked 2x2 tiles/wave) ---------------
// C[M=4096,N=2048] = A[M,K] * BT[N,K]^T      (bf16 in, f32 accumulate)
// Block: 8 waves as 2(M) x 4(N); wave tile 32x32; block tile 64x128.
// MODE 0: A-row r = s*B+b  -> out row b*S+s   (hidden -> q/k/v, bf16 out)
// MODE 1: A-row r = b*S+s  -> out row s*B+b   (o -> final output, f32 out)
template<int MODE, bool OUTBF16>
__global__ __launch_bounds__(256)
void wmma_gemm_kernel(const bf16* __restrict__ A, const bf16* __restrict__ BT,
                      void* __restrict__ outp)
{
    constexpr int Kd = 2048, Nd = 2048;
    const int lane = threadIdx.x & 31;
    const int wave = threadIdx.x >> 5;          // 0..7
    const int wy   = wave >> 2;                 // 0..1  (M)
    const int wx   = wave & 3;                  // 0..3  (N)
    const int m0 = blockIdx.y * 64  + wy * 32;
    const int n0 = blockIdx.x * 128 + wx * 32;
    const int lh = lane >> 4;                   // half-wave select
    const int lm = lane & 15;

    // ISA 7.12.2 16-bit A layout: lanes0-15 K{0-7,16-23}, lanes16-31 K{8-15,24-31}
    const bf16* aptr0 = A  + (size_t)(m0 + lm)      * Kd + lh * 8;
    const bf16* aptr1 = A  + (size_t)(m0 + 16 + lm) * Kd + lh * 8;
    // B layout: lane n = column n; lanes0-15 K{0-7,16-23}, lanes16-31 K{8-15,24-31}
    const bf16* bptr0 = BT + (size_t)(n0 + lm)      * Kd + lh * 8;
    const bf16* bptr1 = BT + (size_t)(n0 + 16 + lm) * Kd + lh * 8;

    v8f acc00 = {0.f,0.f,0.f,0.f,0.f,0.f,0.f,0.f};
    v8f acc01 = acc00, acc10 = acc00, acc11 = acc00;

#pragma unroll 2
    for (int k0 = 0; k0 < Kd; k0 += 32) {
        FragAB a0, a1, b0, b1;
        a0.q[0] = *(const uint4*)(aptr0 + k0);
        a0.q[1] = *(const uint4*)(aptr0 + k0 + 16);
        a1.q[0] = *(const uint4*)(aptr1 + k0);
        a1.q[1] = *(const uint4*)(aptr1 + k0 + 16);
        b0.q[0] = *(const uint4*)(bptr0 + k0);
        b0.q[1] = *(const uint4*)(bptr0 + k0 + 16);
        b1.q[0] = *(const uint4*)(bptr1 + k0);
        b1.q[1] = *(const uint4*)(bptr1 + k0 + 16);
        acc00 = __builtin_amdgcn_wmma_f32_16x16x32_bf16(
                    false, a0.v, false, b0.v, (short)0, acc00, false, false);
        acc01 = __builtin_amdgcn_wmma_f32_16x16x32_bf16(
                    false, a0.v, false, b1.v, (short)0, acc01, false, false);
        acc10 = __builtin_amdgcn_wmma_f32_16x16x32_bf16(
                    false, a1.v, false, b0.v, (short)0, acc10, false, false);
        acc11 = __builtin_amdgcn_wmma_f32_16x16x32_bf16(
                    false, a1.v, false, b1.v, (short)0, acc11, false, false);
    }

    // store 4 tiles; C/D layout: row = vgpr + 8*lh, col = lm
    v8f accs[2][2] = {{acc00, acc01}, {acc10, acc11}};
#pragma unroll
    for (int ti = 0; ti < 2; ++ti) {
#pragma unroll
        for (int tj = 0; tj < 2; ++tj) {
            const int col = n0 + tj * 16 + lm;
#pragma unroll
            for (int vv = 0; vv < 8; ++vv) {
                int r = m0 + ti * 16 + vv + 8 * lh;     // A-row of this element
                int r2;
                if (MODE == 0) r2 = (r & 1) * S_LEN + (r >> 1);          // (s,b)->(b,s)
                else           r2 = ((r & (S_LEN - 1)) << 1) | (r >> 11);// (b,s)->(s,b)
                if (OUTBF16) ((bf16*) outp)[(size_t)r2 * Nd + col] = (bf16)accs[ti][tj][vv];
                else         ((float*)outp)[(size_t)r2 * Nd + col] = accs[ti][tj][vv];
            }
        }
    }
}

// ---------------- causal depthwise conv (K=4) + SiLU ------------------------
__global__ __launch_bounds__(256)
void conv_silu_kernel(const bf16* __restrict__ x, const float* __restrict__ w,
                      bf16* __restrict__ y)
{
    int idx = blockIdx.x * 256 + threadIdx.x;       // over B*S*C
    int c = idx & (C_DIM - 1);
    int s = (idx >> 11) & (S_LEN - 1);
    int b = idx >> 22;
    float acc = 0.f;
#pragma unroll
    for (int i = 0; i < KCONV; ++i) {
        int sp = s - (KCONV - 1) + i;
        if (sp >= 0)
            acc += (float)x[((size_t)(b * S_LEN + sp) << 11) + c] * w[c * KCONV + i];
    }
    float r = acc / (1.f + __expf(-acc));           // SiLU
    y[idx] = (bf16)r;
}

// ---------------- delta-rule recurrence + fused RMSNorm ---------------------
__device__ __forceinline__ float block_reduce_sum(float v, volatile float* red)
{
#pragma unroll
    for (int off = 16; off > 0; off >>= 1) v += __shfl_xor(v, off, 32);
    int w = threadIdx.x >> 5;
    __syncthreads();
    if ((threadIdx.x & 31) == 0) red[w] = v;
    __syncthreads();
    return red[0] + red[1] + red[2] + red[3];
}

__global__ __launch_bounds__(128)
void delta_recurrence_kernel(const bf16* __restrict__ qc, const bf16* __restrict__ kc,
                             const bf16* __restrict__ vc, const float* __restrict__ beta,
                             const float* __restrict__ g_norm, bf16* __restrict__ ob)
{
    const int tid = threadIdx.x;            // state column j (dv)
    const int h   = blockIdx.x & (NH - 1);
    const int b   = blockIdx.x >> 4;

    __shared__ float kL[HD], qL[HD], red[4];

    float S[HD];                            // S[i][tid] in registers
#pragma unroll
    for (int i = 0; i < HD; ++i) S[i] = 0.f;

    const float g = g_norm[tid];
    const float qscale = 0.08838834764831845f;   // 1/sqrt(128)

    for (int t = 0; t < S_LEN; ++t) {
        const size_t base = ((size_t)(b * S_LEN + t) << 11) + h * HD;
        float qv = (float)qc[base + tid];
        float kv = (float)kc[base + tid];
        float vv = (float)vc[base + tid];
        float bt = beta[(size_t)(b * S_LEN + t) * NH + h];

        float ks = block_reduce_sum(kv * kv, red);
        float qs = block_reduce_sum(qv * qv, red);
        float kn = kv * rsqrtf(ks + 1e-12f);
        float qn = qv * rsqrtf(qs + 1e-12f) * qscale;

        kL[tid] = kn; qL[tid] = qn;
        __syncthreads();

        float vp = 0.f;
#pragma unroll
        for (int i = 0; i < HD; ++i) vp += kL[i] * S[i];
        float u = bt * (vv - vp);

        float o = 0.f;
#pragma unroll
        for (int i = 0; i < HD; ++i) { S[i] += kL[i] * u; o += qL[i] * S[i]; }

        float os = block_reduce_sum(o * o, red);
        float scale = rsqrtf(os * (1.f / HD) + 1e-5f) * g;
        ob[base + tid] = (bf16)(o * scale);
        __syncthreads();                    // kL/qL reused next step
    }
}

// ---------------- launch ----------------------------------------------------
extern "C" void kernel_launch(void* const* d_in, const int* in_sizes, int n_in,
                              void* d_out, int out_size, void* d_ws, size_t ws_size,
                              hipStream_t stream)
{
    const float* hs     = (const float*)d_in[0];   // [S,B,H]
    const float* Wq     = (const float*)d_in[1];
    const float* Wk     = (const float*)d_in[2];
    const float* Wv     = (const float*)d_in[3];
    const float* Wb     = (const float*)d_in[4];
    const float* conv_q = (const float*)d_in[5];
    const float* conv_k = (const float*)d_in[6];
    const float* conv_v = (const float*)d_in[7];
    const float* g_norm = (const float*)d_in[8];
    const float* Wo     = (const float*)d_in[9];
    float* out = (float*)d_out;                    // [S,B,H] fp32

    const size_t ROWS = (size_t)S_LEN * B_SZ;      // 4096
    const size_t HS_N = ROWS * H_DIM;              // 8.39M
    const size_t W_N  = (size_t)H_DIM * C_DIM;     // 4.19M
    const size_t ACT_N = ROWS * C_DIM;             // 8.39M

    char* p = (char*)d_ws;
    bf16* hsb = (bf16*)p; p += HS_N * sizeof(bf16);
    bf16* WqT = (bf16*)p; p += W_N * sizeof(bf16);
    bf16* WkT = (bf16*)p; p += W_N * sizeof(bf16);
    bf16* WvT = (bf16*)p; p += W_N * sizeof(bf16);
    bf16* WoT = (bf16*)p; p += W_N * sizeof(bf16);
    bf16* qb  = (bf16*)p; p += ACT_N * sizeof(bf16);
    bf16* kb  = (bf16*)p; p += ACT_N * sizeof(bf16);
    bf16* vb  = (bf16*)p; p += ACT_N * sizeof(bf16);
    bf16* qcb = (bf16*)p; p += ACT_N * sizeof(bf16);
    bf16* kcb = (bf16*)p; p += ACT_N * sizeof(bf16);
    bf16* vcb = (bf16*)p; p += ACT_N * sizeof(bf16);
    bf16* obf = (bf16*)p; p += ACT_N * sizeof(bf16);
    float* beta = (float*)p; p += ROWS * NH * sizeof(float);

    // 1. conversions
    convert_bf16_kernel<<<(int)((HS_N + 255) / 256), 256, 0, stream>>>(hs, hsb, (int)HS_N);
    const int tgrid = (int)(W_N / 256);
    transpose_convert_kernel<<<tgrid, 256, 0, stream>>>(Wq, WqT);
    transpose_convert_kernel<<<tgrid, 256, 0, stream>>>(Wk, WkT);
    transpose_convert_kernel<<<tgrid, 256, 0, stream>>>(Wv, WvT);
    transpose_convert_kernel<<<tgrid, 256, 0, stream>>>(Wo, WoT);

    // 2. beta projection (fp32)
    beta_kernel<<<(int)ROWS, 256, 0, stream>>>(hs, Wb, beta);

    // 3. QKV projections via WMMA
    dim3 ggrid(C_DIM / 128, (int)(ROWS / 64));
    wmma_gemm_kernel<0, true><<<ggrid, 256, 0, stream>>>(hsb, WqT, qb);
    wmma_gemm_kernel<0, true><<<ggrid, 256, 0, stream>>>(hsb, WkT, kb);
    wmma_gemm_kernel<0, true><<<ggrid, 256, 0, stream>>>(hsb, WvT, vb);

    // 4. causal conv + SiLU
    const int cgrid = (int)(ACT_N / 256);
    conv_silu_kernel<<<cgrid, 256, 0, stream>>>(qb, conv_q, qcb);
    conv_silu_kernel<<<cgrid, 256, 0, stream>>>(kb, conv_k, kcb);
    conv_silu_kernel<<<cgrid, 256, 0, stream>>>(vb, conv_v, vcb);

    // 5. delta-rule recurrence (fused l2norm, q-scale, RMSNorm, g_norm)
    delta_recurrence_kernel<<<B_SZ * NH, 128, 0, stream>>>(qcb, kcb, vcb, beta,
                                                           g_norm, obf);

    // 6. output projection via WMMA, fp32 out in [s,b,H] order
    wmma_gemm_kernel<1, false><<<ggrid, 256, 0, stream>>>(obf, WoT, out);
}
